// HetFormer_60206851555970
// MI455X (gfx1250) — matrix operand order repaired and checked by reading, not compiled, not run on hardware
//
#include <hip/hip_runtime.h>
#include <math.h>

typedef unsigned short u16;
typedef __attribute__((ext_vector_type(16))) __bf16 bf16x16;
typedef __attribute__((ext_vector_type(8)))  float  f32x8;

union Frag16 { bf16x16 v; unsigned u[8]; };

__device__ __forceinline__ u16 f2bf(float f) {
    union { __bf16 h; u16 s; } cv; cv.h = (__bf16)f; return cv.s;
}

#define TM 64
#define TN 64
#define LPAD 40   // u16 row stride: 80B -> uint4 LDS ops aligned, dword frag reads aligned

// ---------------------------------------------------------------------------
// Generic batched WMMA GEMM with TRANSPOSED B operand storage:
//   out[M,N] = scale*(A_bf16[M,K] @ B + bias),  B supplied as Wt[N][K] bf16.
// REQUIRES: M % 64 == 0, K % 32 == 0 (true at every call site).
// Both tiles are contiguous uint4 copies into LDS (no transpose scatter).
// Out-of-range Wt rows (n >= N) are zero-filled in LDS; stores guarded nG < N.
// 8 waves; each wave owns two 16x16 accumulators sharing one B fragment.
// obT: write bf16 output transposed (outB[n*ldc + m]) -- used for V^T.
// z-axis: zo=z/zdiv, zi=z%zdiv; base += zo*Outer + zi*Inner (A, W, out).
// ---------------------------------------------------------------------------
__global__ __launch_bounds__(256) void k_gemm(
    const u16* __restrict__ A, int lda, long aO, long aI,
    const u16* __restrict__ Wt, int ldw, long wO, long wI,
    const float* __restrict__ bias,
    float* __restrict__ outF, u16* __restrict__ outB, int ldc, long oO, long oI,
    int M, int N, int K, int zdiv, float scale, int relu, int obT)
{
    __shared__ u16 As[TM * LPAD];
    __shared__ u16 Bs[TN * LPAD];
    int z = blockIdx.z, zo = z / zdiv, zi = z % zdiv;
    A  += zo * aO + zi * aI;
    Wt += zo * wO + zi * wI;
    long obase = zo * oO + zi * oI;
    int n0 = blockIdx.x * TN, m0 = blockIdx.y * TM;
    int tid = threadIdx.x, lane = tid & 31, wave = tid >> 5;
    int wr = wave >> 2, wc = wave & 3, half = lane >> 4, l16 = lane & 15;

    // staging indices: 64 rows x 32 cols each, 8 contiguous elems per thread
    int sr = tid >> 2, sc = (tid & 3) * 8;
    bool bvalid = (n0 + sr < N);

    f32x8 acc0 = {0.f,0.f,0.f,0.f,0.f,0.f,0.f,0.f};
    f32x8 acc1 = {0.f,0.f,0.f,0.f,0.f,0.f,0.f,0.f};

    for (int k0 = 0; k0 < K; k0 += 32) {
        __syncthreads();
        *(uint4*)&As[sr * LPAD + sc] =
            *(const uint4*)&A[(long)(m0 + sr) * lda + k0 + sc];
        uint4 bd = {0u, 0u, 0u, 0u};
        if (bvalid) bd = *(const uint4*)&Wt[(long)(n0 + sr) * ldw + k0 + sc];
        *(uint4*)&Bs[sr * LPAD + sc] = bd;
        __syncthreads();
        Frag16 fa0, fa1, fb;
        int mrow = wr * 16 + l16;
        int ncol = wc * 16 + l16;
#pragma unroll
        for (int v = 0; v < 8; ++v) {
            // ISA 7.12.2: A(16x32 bf16) lane m=l16, half selects K group
            int ka = (v >> 2) * 16 + half * 8 + (v & 3) * 2;
            fa0.u[v] = *(const unsigned*)&As[mrow * LPAD + ka];
            fa1.u[v] = *(const unsigned*)&As[(mrow + 32) * LPAD + ka];
            // B(32x16): same K order as A (half selects K group), 2 K per VGPR
            int kb = (v >> 2) * 16 + half * 8 + (v & 3) * 2;
            fb.u[v] = *(const unsigned*)&Bs[ncol * LPAD + kb];
        }
        acc0 = __builtin_amdgcn_wmma_f32_16x16x32_bf16(false, fa0.v, false, fb.v,
                                                       (short)0, acc0, false, false);
        acc1 = __builtin_amdgcn_wmma_f32_16x16x32_bf16(false, fa1.v, false, fb.v,
                                                       (short)0, acc1, false, false);
    }
    int nG = n0 + wc * 16 + l16;
    float bv = (bias && nG < N) ? bias[nG] : 0.f;
#pragma unroll
    for (int a = 0; a < 2; ++a) {
        f32x8 acc = a ? acc1 : acc0;
#pragma unroll
        for (int r = 0; r < 8; ++r) {
            int mG = m0 + wr * 16 + a * 32 + r + half * 8;  // D: n=lane%16, m=r+8*half
            if (nG < N) {
                float x = (acc[r] + bv) * scale;
                if (relu) x = fmaxf(x, 0.f);
                if (outF) outF[obase + (long)mG * ldc + nG] = x;
                if (outB) {
                    long oi = obT ? obase + (long)nG * ldc + mG
                                  : obase + (long)mG * ldc + nG;
                    outB[oi] = f2bf(x);
                }
            }
        }
    }
}

// ---------------------------------------------------------------------------
// WMMA A·B^T over K=32 (one wmma step). out[m,n] = dot(A[m,:32], B[n,:32])
// REQUIRES: M % 64 == 0, N % 64 == 0 (always 256x256 at call sites).
// mode 0: scores:  out = acc (+ cw*pair[m,n or n,m] + cb)
// mode 1: pair:    out[m,n] += pw * mask(m,n) * acc        (out aliases pair)
// mode 2: gate:    out = mask * sigmoid(sw*pair[m,n] + sb + acc)
// ---------------------------------------------------------------------------
__global__ __launch_bounds__(256) void k_abt(
    const u16* __restrict__ A, int lda, long aO, long aI,
    const u16* __restrict__ Bm, int ldb, long bO, long bI,
    float* __restrict__ out, int ldo, long oO, long oI,
    const float* __restrict__ pair, int ldp, long pO, int pT,
    const float* cwp, const float* cbp, int cidx,
    const float* pwp, int pwidx,
    const float* swp, const float* sbp,
    const int* mlenp, const int* nlenp,
    int M, int N, int zdiv, int mode)
{
    __shared__ u16 As[TM * LPAD];
    __shared__ u16 Bs[TN * LPAD];
    int z = blockIdx.z, zo = z / zdiv, zi = z % zdiv;
    A  += zo * aO + zi * aI;
    Bm += zo * bO + zi * bI;
    long obase = zo * oO + zi * oI;
    const float* pr = pair ? pair + zo * pO : (const float*)0;
    int n0 = blockIdx.x * TN, m0 = blockIdx.y * TM;
    int tid = threadIdx.x, lane = tid & 31, wave = tid >> 5;
    int wr = wave >> 2, wc = wave & 3, half = lane >> 4, l16 = lane & 15;

    // A: 64x32, B rows already K-contiguous -> both tiles are pure 16B copies
    {
        int r = tid >> 2, c = (tid & 3) * 8;
        *(uint4*)&As[r * LPAD + c] = *(const uint4*)&A[(long)(m0 + r) * lda + c];
        *(uint4*)&Bs[r * LPAD + c] = *(const uint4*)&Bm[(long)(n0 + r) * ldb + c];
    }
    __syncthreads();
    Frag16 fa0, fa1, fb;
    int mrow = wr * 16 + l16, ncol = wc * 16 + l16;
#pragma unroll
    for (int v = 0; v < 8; ++v) {
        int ka = (v >> 2) * 16 + half * 8 + (v & 3) * 2;
        fa0.u[v] = *(const unsigned*)&As[mrow * LPAD + ka];
        fa1.u[v] = *(const unsigned*)&As[(mrow + 32) * LPAD + ka];
        int kb = (v >> 2) * 16 + half * 8 + (v & 3) * 2;
        fb.u[v] = *(const unsigned*)&Bs[ncol * LPAD + kb];
    }
    f32x8 acc0 = {0.f,0.f,0.f,0.f,0.f,0.f,0.f,0.f};
    f32x8 acc1 = {0.f,0.f,0.f,0.f,0.f,0.f,0.f,0.f};
    acc0 = __builtin_amdgcn_wmma_f32_16x16x32_bf16(false, fa0.v, false, fb.v,
                                                   (short)0, acc0, false, false);
    acc1 = __builtin_amdgcn_wmma_f32_16x16x32_bf16(false, fa1.v, false, fb.v,
                                                   (short)0, acc1, false, false);
    int nG = n0 + wc * 16 + l16;
    int ml = mlenp ? *mlenp : M;
    int nl = nlenp ? *nlenp : N;
    float cw = cwp ? cwp[cidx] : 0.f, cb = cbp ? cbp[cidx] : 0.f;
    float pw = pwp ? pwp[pwidx] : 0.f;
    float sw = swp ? *swp : 0.f,    sb = sbp ? *sbp : 0.f;
#pragma unroll
    for (int a = 0; a < 2; ++a) {
        f32x8 acc = a ? acc1 : acc0;
#pragma unroll
        for (int r = 0; r < 8; ++r) {
            int mG = m0 + wr * 16 + a * 32 + r + half * 8;
            float x = acc[r];
            long oi = obase + (long)mG * ldo + nG;
            if (mode == 0) {
                if (pr) x += cw * pr[pT ? (long)nG * ldp + mG : (long)mG * ldp + nG] + cb;
                out[oi] = x;
            } else if (mode == 1) {
                bool msk = (mG < ml) && (nG < nl);
                out[oi] += pw * (msk ? x : 0.f);
            } else {
                bool msk = (mG < ml) && (nG < nl);
                float g = 0.f;
                if (msk) {
                    float t = sw * pr[(long)mG * ldp + nG] + sb + x;
                    g = 1.f / (1.f + __expf(-t));
                }
                out[oi] = g;
            }
        }
    }
}

// ---------------------------------------------------------------------------
// Rowwise masked softmax over 256 cols; writes bf16 probs.
// ---------------------------------------------------------------------------
__global__ __launch_bounds__(256) void k_softmax(
    const float* __restrict__ S, u16* __restrict__ P,
    const int* qlenp, const int* klenp, int postmask)
{
    __shared__ float red[256];
    int row = blockIdx.x, q = row & 255, t = threadIdx.x;
    long base = (long)row * 256;
    int qlen = *qlenp, klen = *klenp;
    if (q >= qlen || klen <= 0) {
        P[base + t] = postmask ? (u16)0 : f2bf(1.f / 256.f);
        return;
    }
    float v = (t < klen) ? S[base + t] : -3.4e38f;
    red[t] = v; __syncthreads();
    for (int s = 128; s > 0; s >>= 1) { if (t < s) red[t] = fmaxf(red[t], red[t + s]); __syncthreads(); }
    float m = red[0]; __syncthreads();
    float e = (t < klen) ? __expf(v - m) : 0.f;
    red[t] = e; __syncthreads();
    for (int s = 128; s > 0; s >>= 1) { if (t < s) red[t] += red[t + s]; __syncthreads(); }
    float sum = red[0];
    P[base + t] = f2bf(e / sum);
}

// residual add + LayerNorm over C=256; refreshes f32 state and bf16 shadow
__global__ __launch_bounds__(256) void k_add_ln(
    float* __restrict__ X, u16* __restrict__ Xbf, const float* __restrict__ U,
    const float* __restrict__ g, const float* __restrict__ b)
{
    __shared__ float red[256];
    long base = (long)blockIdx.x * 256; int t = threadIdx.x;
    float y = X[base + t] + U[base + t];
    red[t] = y; __syncthreads();
    for (int s = 128; s > 0; s >>= 1) { if (t < s) red[t] += red[t + s]; __syncthreads(); }
    float mu = red[0] * (1.f / 256.f); __syncthreads();
    float d = y - mu;
    red[t] = d * d; __syncthreads();
    for (int s = 128; s > 0; s >>= 1) { if (t < s) red[t] += red[t + s]; __syncthreads(); }
    float var = red[0] * (1.f / 256.f); __syncthreads();
    float o = g[t] * d * rsqrtf(var + 1e-6f) + b[t];
    X[base + t] = o; Xbf[base + t] = f2bf(o);
}

// embedding: enc = valid * (x[5] @ emb_w + emb_b + sinusoid(pos))
__global__ __launch_bounds__(256) void k_embed(
    const float* __restrict__ X, const float* __restrict__ We,
    const float* __restrict__ Be, const int* lenp,
    float* __restrict__ E, u16* __restrict__ Ebf)
{
    int row = blockIdx.x, pos = row & 255, t = threadIdx.x;
    int len = *lenp;
    const float* xr = X + (long)row * 5;
    float acc = Be[t];
#pragma unroll
    for (int f = 0; f < 5; ++f) acc += xr[f] * We[f * 256 + t];
    float div = __expf(-logf(10000.f) * (2.f * (float)(t >> 1)) * (1.f / 256.f));
    float ang = (float)pos * div;
    acc += (t & 1) ? __cosf(ang) : __sinf(ang);
    float o = (pos < len) ? acc : 0.f;
    long base = (long)row * 256;
    E[base + t] = o; Ebf[base + t] = f2bf(o);
}

__global__ void k_pair_init(const float* __restrict__ bpp, float* __restrict__ pair, int n) {
    int i = blockIdx.x * 256 + threadIdx.x;
    if (i >= n) return;
    float p = bpp[i];
    pair[i] = logf(p + 1e-3f) - logf(1.f - p + 1e-3f);
}

// f32 [nm][K][N] -> bf16 transposed [nm][N][K]  (dst index coalesced)
__global__ void k_cvt_t(const float* __restrict__ s, u16* __restrict__ d,
                        int K, int N, long total)
{
    long i = (long)blockIdx.x * 256 + threadIdx.x;
    if (i >= total) return;
    long kn = (long)K * N;
    long m = i / kn, rem = i - m * kn;
    int n = (int)(rem / K), k = (int)(rem - (long)n * K);
    d[i] = f2bf(s[m * kn + (long)k * N + n]);
}

// row/col sums of gate matrix [256x256] per sample
__global__ __launch_bounds__(256) void k_rowcol(const float* __restrict__ G,
                                                float* __restrict__ gu, float* __restrict__ ge)
{
    int b = blockIdx.x, t = threadIdx.x;
    const float* g = G + (long)b * 65536;
    float s = 0.f;
    for (int e = 0; e < 256; ++e) s += g[t * 256 + e];
    gu[b * 256 + t] = s;
    float s2 = 0.f;
    for (int u = 0; u < 256; ++u) s2 += g[u * 256 + t];
    ge[b * 256 + t] = s2;
}

// per-position scalar gate: sigmoid(enc . gate_w + gb), masked
__global__ __launch_bounds__(256) void k_gatevec(
    const float* __restrict__ uenc, const float* __restrict__ eenc,
    const float* __restrict__ gw, const float* __restrict__ gb,
    const int* ulp, const int* elp, float* __restrict__ ug, float* __restrict__ eg)
{
    __shared__ float red[256];
    int row = blockIdx.x, which = blockIdx.y, t = threadIdx.x;
    const float* enc = which ? eenc : uenc;
    int len = which ? *elp : *ulp;
    red[t] = enc[(long)row * 256 + t] * gw[t];
    __syncthreads();
    for (int s = 128; s > 0; s >>= 1) { if (t < s) red[t] += red[t + s]; __syncthreads(); }
    if (t == 0) {
        int pos = row & 255;
        float o = (pos < len) ? 1.f / (1.f + __expf(-(red[0] + gb[0]))) : 0.f;
        (which ? eg : ug)[row] = o;
    }
}

// fv[b, off + c] = sum_r w[b,r] * enc[b,r,c]
__global__ __launch_bounds__(256) void k_wsum(const float* __restrict__ w,
                                              const float* __restrict__ enc,
                                              float* __restrict__ fv, int off)
{
    int b = blockIdx.x, t = threadIdx.x;
    const float* wr = w + b * 256;
    const float* e = enc + (long)b * 65536;
    float acc = 0.f;
    for (int r = 0; r < 256; ++r) acc += wr[r] * e[r * 256 + t];
    fv[(long)b * 1057 + off + t] = acc;
}

__global__ void k_tail(const float* __restrict__ oh, const float* __restrict__ he,
                       float* __restrict__ fv)
{
    int b = blockIdx.x, t = threadIdx.x;
    if (t < 32) fv[(long)b * 1057 + 1024 + t] = oh[b * 32 + t];
    if (t == 32) fv[(long)b * 1057 + 1056] = he[b];
}

// final head: LN(1057) -> relu(@od_w[1057x256]) -> @ow_w[256x3]
__global__ __launch_bounds__(256) void k_head(
    const float* __restrict__ fv, const float* __restrict__ ons, const float* __restrict__ onb,
    const float* __restrict__ odw, const float* __restrict__ odb,
    const float* __restrict__ oww, const float* __restrict__ owb, float* __restrict__ out)
{
    __shared__ float nv[1057];
    __shared__ float hid[256];
    __shared__ float red[256];
    int b = blockIdx.x, t = threadIdx.x;
    const float* v = fv + (long)b * 1057;
    float ls = 0.f;
    for (int j = t; j < 1057; j += 256) ls += v[j];
    red[t] = ls; __syncthreads();
    for (int s = 128; s > 0; s >>= 1) { if (t < s) red[t] += red[t + s]; __syncthreads(); }
    float mu = red[0] / 1057.f; __syncthreads();
    float lq = 0.f;
    for (int j = t; j < 1057; j += 256) { float d = v[j] - mu; lq += d * d; }
    red[t] = lq; __syncthreads();
    for (int s = 128; s > 0; s >>= 1) { if (t < s) red[t] += red[t + s]; __syncthreads(); }
    float var = red[0] / 1057.f; __syncthreads();
    float inv = rsqrtf(var + 1e-6f);
    for (int j = t; j < 1057; j += 256) nv[j] = ons[j] * (v[j] - mu) * inv + onb[j];
    __syncthreads();
    float h = odb[t];
    for (int j = 0; j < 1057; ++j) h += nv[j] * odw[j * 256 + t];
    hid[t] = fmaxf(h, 0.f);
    __syncthreads();
    if (t < 3) {
        float o = owb[t];
        for (int j = 0; j < 256; ++j) o += hid[j] * oww[j * 3 + t];
        out[b * 3 + t] = o;
    }
}

// ===========================================================================
extern "C" void kernel_launch(void* const* d_in, const int* in_sizes, int n_in,
                              void* d_out, int out_size, void* d_ws, size_t ws_size,
                              hipStream_t stream)
{
    (void)in_sizes; (void)n_in; (void)out_size; (void)ws_size;
    const int Bc = 16, Hh = 8, NBk = 4, NRr = 2;
    const long SE = 65536;           // 256*256
    const float qscale = 0.17677669529663687f;   // 1/sqrt(32)

    const float* in_u    = (const float*)d_in[0];
    const float* in_e    = (const float*)d_in[1];
    const float* in_bpp  = (const float*)d_in[2];
    const float* in_oh   = (const float*)d_in[3];
    const float* in_he   = (const float*)d_in[4];
    const int*   p_ulen  = (const int*)d_in[5];
    const int*   p_elen  = (const int*)d_in[6];
    const float* emb_w   = (const float*)d_in[7];
    const float* emb_b   = (const float*)d_in[8];
    const float* ca_wq   = (const float*)d_in[9];
    const float* ca_wq_b = (const float*)d_in[10];
    const float* ca_wk   = (const float*)d_in[11];
    const float* ca_wk_b = (const float*)d_in[12];
    const float* ca_wv   = (const float*)d_in[13];
    const float* ca_wv_b = (const float*)d_in[14];
    const float* ca_wo   = (const float*)d_in[15];
    const float* ca_wo_b = (const float*)d_in[16];
    const float* ca_bw   = (const float*)d_in[17];
    const float* ca_bb   = (const float*)d_in[18];
    const float* sa_wq   = (const float*)d_in[19];
    const float* sa_wq_b = (const float*)d_in[20];
    const float* sa_wk   = (const float*)d_in[21];
    const float* sa_wk_b = (const float*)d_in[22];
    const float* sa_wv   = (const float*)d_in[23];
    const float* sa_wv_b = (const float*)d_in[24];
    const float* sa_wo   = (const float*)d_in[25];
    const float* sa_wo_b = (const float*)d_in[26];
    const float* mlp_iw  = (const float*)d_in[27];
    const float* mlp_ib  = (const float*)d_in[28];
    const float* mlp_ow  = (const float*)d_in[29];
    const float* mlp_ob  = (const float*)d_in[30];
    const float* cn_s = (const float*)d_in[31];
    const float* cn_b = (const float*)d_in[32];
    const float* sn_s = (const float*)d_in[33];
    const float* sn_b = (const float*)d_in[34];
    const float* mn_s = (const float*)d_in[35];
    const float* mn_b = (const float*)d_in[36];
    const float* pair_w = (const float*)d_in[37];
    const float* pu_w   = (const float*)d_in[38];
    const float* pu_b   = (const float*)d_in[39];
    const float* so_bw  = (const float*)d_in[40];
    const float* so_bb  = (const float*)d_in[41];
    const float* so_w   = (const float*)d_in[42];
    const float* so_b   = (const float*)d_in[43];
    const float* gate_w = (const float*)d_in[44];
    const float* gate_b = (const float*)d_in[45];
    const float* on_s = (const float*)d_in[46];
    const float* on_b = (const float*)d_in[47];
    const float* od_w = (const float*)d_in[48];
    const float* od_b = (const float*)d_in[49];
    const float* ow_w = (const float*)d_in[50];
    const float* ow_b = (const float*)d_in[51];

    char* ws = (char*)d_ws;
    size_t off = 0;
    auto alloc = [&](size_t bytes) -> char* {
        char* p = ws + off;
        off = (off + bytes + 255) & ~(size_t)255;
        return p;
    };
    // transpose-convert: src f32 [nm][K][N] -> dst bf16 [nm][N][K]
    auto cvtT = [&](const float* src, int K, int N, int nm) -> u16* {
        long total = (long)K * N * nm;
        u16* d = (u16*)alloc((size_t)total * 2);
        k_cvt_t<<<dim3((unsigned)((total + 255) / 256)), 256, 0, stream>>>(src, d, K, N, total);
        return d;
    };

    // ---- bf16 transposed weight shadows [N][K] (~8.5 MB, built once/call) ----
    u16* bq_ca = cvtT(ca_wq, 256, 256, NBk);
    u16* bk_ca = cvtT(ca_wk, 256, 256, NBk);
    u16* bv_ca = cvtT(ca_wv, 256, 256, NBk);
    u16* bo_ca = cvtT(ca_wo, 256, 256, NBk);
    u16* bq_sa = cvtT(sa_wq, 256, 256, NBk);
    u16* bk_sa = cvtT(sa_wk, 256, 256, NBk);
    u16* bv_sa = cvtT(sa_wv, 256, 256, NBk);
    u16* bo_sa = cvtT(sa_wo, 256, 256, NBk);
    u16* bmlp_i = cvtT(mlp_iw, 256, 1024, NBk);   // [1024][256] per block
    u16* bmlp_o = cvtT(mlp_ow, 1024, 256, NBk);   // [256][1024] per block
    u16* bpu = cvtT(pu_w, 256, 32, NBk);          // [32][256] per block
    u16* bso = cvtT(so_w, 256, 32, 1);            // [32][256]

    // ---- state / scratch (~95 MB) ----
    float* u_enc = (float*)alloc((size_t)Bc * SE * 4);
    float* e_enc = (float*)alloc((size_t)Bc * SE * 4);
    float* pair  = (float*)alloc((size_t)Bc * SE * 4);
    u16* u_bf = (u16*)alloc((size_t)Bc * SE * 2);
    u16* e_bf = (u16*)alloc((size_t)Bc * SE * 2);
    u16* qb = (u16*)alloc((size_t)Bc * SE * 2);
    u16* kb = (u16*)alloc((size_t)Bc * SE * 2);
    u16* vb = (u16*)alloc((size_t)Bc * SE * 2);   // V^T layout [hd][e] per sample
    u16* q2 = (u16*)alloc((size_t)Bc * SE * 2);
    u16* k2 = (u16*)alloc((size_t)Bc * SE * 2);
    u16* v2 = (u16*)alloc((size_t)Bc * SE * 2);   // V^T layout
    float* scores = (float*)alloc((size_t)Bc * Hh * SE * 4);
    u16* probs = (u16*)alloc((size_t)Bc * Hh * SE * 2);
    u16* ctx = (u16*)alloc((size_t)Bc * SE * 2);
    float* upd_u = (float*)alloc((size_t)Bc * SE * 4);
    float* upd_e = (float*)alloc((size_t)Bc * SE * 4);
    u16* mlp_h = (u16*)alloc((size_t)Bc * 256 * 1024 * 2);
    u16* pub = (u16*)alloc((size_t)Bc * 256 * 32 * 2);
    u16* peb = (u16*)alloc((size_t)Bc * 256 * 32 * 2);
    u16* sub_ = (u16*)alloc((size_t)Bc * 256 * 32 * 2);
    u16* seb = (u16*)alloc((size_t)Bc * 256 * 32 * 2);
    float* gu  = (float*)alloc((size_t)Bc * 256 * 4);
    float* ge  = (float*)alloc((size_t)Bc * 256 * 4);
    float* ugv = (float*)alloc((size_t)Bc * 256 * 4);
    float* egv = (float*)alloc((size_t)Bc * 256 * 4);
    float* fv  = (float*)alloc((size_t)Bc * 1057 * 4);

    auto gemm = [&](const u16* A, int lda, long aO, long aI,
                    const u16* Wt, int ldw, long wO, long wI,
                    const float* bias, float* oF, u16* oB, int ldc, long oO, long oI,
                    int M, int N, int K, int Z, int zdiv, float scale, int relu, int obT) {
        k_gemm<<<dim3((N + 63) / 64, (M + 63) / 64, Z), 256, 0, stream>>>(
            A, lda, aO, aI, Wt, ldw, wO, wI, bias, oF, oB, ldc, oO, oI,
            M, N, K, zdiv, scale, relu, obT);
    };

    // ---- init ----
    k_embed<<<Bc * 256, 256, 0, stream>>>(in_u, emb_w, emb_b, p_ulen, u_enc, u_bf);
    k_embed<<<Bc * 256, 256, 0, stream>>>(in_e, emb_w, emb_b, p_elen, e_enc, e_bf);
    k_pair_init<<<(Bc * (int)SE + 255) / 256, 256, 0, stream>>>(in_bpp, pair, Bc * (int)SE);

    auto attn_self = [&](u16* xbf, float* xenc, int i,
                         const float* bq, const float* bk, const float* bv, const float* bo,
                         const int* lenp, const float* lns, const float* lnb) {
        long WQ = (long)i * SE;
        gemm(xbf, 256, SE, 0, bq_sa + WQ, 256, 0, 0, bq + (long)i * 256, nullptr, qb, 256, SE, 0, 256, 256, 256, Bc, 1, qscale, 0, 0);
        gemm(xbf, 256, SE, 0, bk_sa + WQ, 256, 0, 0, bk + (long)i * 256, nullptr, kb, 256, SE, 0, 256, 256, 256, Bc, 1, 1.f, 0, 0);
        gemm(xbf, 256, SE, 0, bv_sa + WQ, 256, 0, 0, bv + (long)i * 256, nullptr, vb, 256, SE, 0, 256, 256, 256, Bc, 1, 1.f, 0, 1);
        k_abt<<<dim3(4, 4, Bc * Hh), 256, 0, stream>>>(
            qb, 256, SE, 32, kb, 256, SE, 32, scores, 256, (long)Hh * SE, SE,
            nullptr, 0, 0, 0, nullptr, nullptr, 0, nullptr, 0, nullptr, nullptr,
            nullptr, nullptr, 256, 256, Hh, 0);
        k_softmax<<<Bc * Hh * 256, 256, 0, stream>>>(scores, probs, lenp, lenp, 0);
        gemm(probs, 256, (long)Hh * SE, SE, vb, 256, SE, 8192, nullptr, nullptr, ctx, 256, SE, 32, 256, 32, 256, Bc * Hh, Hh, 1.f, 0, 0);
        gemm(ctx, 256, SE, 0, bo_sa + WQ, 256, 0, 0, bo + (long)i * 256, upd_u, nullptr, 256, SE, 0, 256, 256, 256, Bc, 1, 1.f, 0, 0);
        k_add_ln<<<Bc * 256, 256, 0, stream>>>(xenc, xbf, upd_u, lns + (long)i * 256, lnb + (long)i * 256);
    };
    auto mlp = [&](u16* xbf, float* xenc, int i) {
        gemm(xbf, 256, SE, 0, bmlp_i + (long)i * 262144, 256, 0, 0, mlp_ib + (long)i * 1024,
             nullptr, mlp_h, 1024, 262144, 0, 256, 1024, 256, Bc, 1, 1.f, 1, 0);
        gemm(mlp_h, 1024, 262144, 0, bmlp_o + (long)i * 262144, 1024, 0, 0, mlp_ob + (long)i * 256,
             upd_u, nullptr, 256, SE, 0, 256, 256, 1024, Bc, 1, 1.f, 0, 0);
        k_add_ln<<<Bc * 256, 256, 0, stream>>>(xenc, xbf, upd_u, mn_s + (long)i * 256, mn_b + (long)i * 256);
    };

    for (int r = 0; r < NRr; ++r) {
        for (int i = 0; i < NBk; ++i) {
            long WQ = (long)i * SE;
            // ===== cross attention =====
            gemm(u_bf, 256, SE, 0, bq_ca + WQ, 256, 0, 0, ca_wq_b + (long)i * 256, nullptr, qb, 256, SE, 0, 256, 256, 256, Bc, 1, qscale, 0, 0);
            gemm(e_bf, 256, SE, 0, bq_ca + WQ, 256, 0, 0, ca_wq_b + (long)i * 256, nullptr, q2, 256, SE, 0, 256, 256, 256, Bc, 1, qscale, 0, 0);
            gemm(u_bf, 256, SE, 0, bk_ca + WQ, 256, 0, 0, ca_wk_b + (long)i * 256, nullptr, kb, 256, SE, 0, 256, 256, 256, Bc, 1, 1.f, 0, 0);
            gemm(e_bf, 256, SE, 0, bk_ca + WQ, 256, 0, 0, ca_wk_b + (long)i * 256, nullptr, k2, 256, SE, 0, 256, 256, 256, Bc, 1, 1.f, 0, 0);
            gemm(u_bf, 256, SE, 0, bv_ca + WQ, 256, 0, 0, ca_wv_b + (long)i * 256, nullptr, vb, 256, SE, 0, 256, 256, 256, Bc, 1, 1.f, 0, 1);
            gemm(e_bf, 256, SE, 0, bv_ca + WQ, 256, 0, 0, ca_wv_b + (long)i * 256, nullptr, v2, 256, SE, 0, 256, 256, 256, Bc, 1, 1.f, 0, 1);
            // u->e direction:  S = q_u . k_e^T + (cw*pair + cb)
            k_abt<<<dim3(4, 4, Bc * Hh), 256, 0, stream>>>(
                qb, 256, SE, 32, k2, 256, SE, 32, scores, 256, (long)Hh * SE, SE,
                pair, 256, SE, 0, ca_bw, ca_bb, i, nullptr, 0, nullptr, nullptr,
                nullptr, nullptr, 256, 256, Hh, 0);
            k_softmax<<<Bc * Hh * 256, 256, 0, stream>>>(scores, probs, p_ulen, p_elen, 1);
            gemm(probs, 256, (long)Hh * SE, SE, v2, 256, SE, 8192, nullptr, nullptr, ctx, 256, SE, 32, 256, 32, 256, Bc * Hh, Hh, 1.f, 0, 0);
            gemm(ctx, 256, SE, 0, bo_ca + WQ, 256, 0, 0, ca_wo_b + (long)i * 256, upd_u, nullptr, 256, SE, 0, 256, 256, 256, Bc, 1, 1.f, 0, 0);
            // e->u direction:  S = q_e . k_u^T + (cw*pair^T + cb)
            k_abt<<<dim3(4, 4, Bc * Hh), 256, 0, stream>>>(
                q2, 256, SE, 32, kb, 256, SE, 32, scores, 256, (long)Hh * SE, SE,
                pair, 256, SE, 1, ca_bw, ca_bb, i, nullptr, 0, nullptr, nullptr,
                nullptr, nullptr, 256, 256, Hh, 0);
            k_softmax<<<Bc * Hh * 256, 256, 0, stream>>>(scores, probs, p_elen, p_ulen, 1);
            gemm(probs, 256, (long)Hh * SE, SE, vb, 256, SE, 8192, nullptr, nullptr, ctx, 256, SE, 32, 256, 32, 256, Bc * Hh, Hh, 1.f, 0, 0);
            gemm(ctx, 256, SE, 0, bo_ca + WQ, 256, 0, 0, ca_wo_b + (long)i * 256, upd_e, nullptr, 256, SE, 0, 256, 256, 256, Bc, 1, 1.f, 0, 0);
            k_add_ln<<<Bc * 256, 256, 0, stream>>>(u_enc, u_bf, upd_u, cn_s + (long)i * 256, cn_b + (long)i * 256);
            k_add_ln<<<Bc * 256, 256, 0, stream>>>(e_enc, e_bf, upd_e, cn_s + (long)i * 256, cn_b + (long)i * 256);
            // ===== self attention =====
            attn_self(u_bf, u_enc, i, sa_wq_b, sa_wk_b, sa_wv_b, sa_wo_b, p_ulen, sn_s, sn_b);
            attn_self(e_bf, e_enc, i, sa_wq_b, sa_wk_b, sa_wv_b, sa_wo_b, p_elen, sn_s, sn_b);
            // ===== MLP =====
            mlp(u_bf, u_enc, i);
            mlp(e_bf, e_enc, i);
            // ===== pair update: pair += pair_w[i] * mask * (p_u @ p_e^T) =====
            gemm(u_bf, 256, SE, 0, bpu + (long)i * 8192, 256, 0, 0, pu_b + (long)i * 32, nullptr, pub, 32, 8192, 0, 256, 32, 256, Bc, 1, 1.f, 0, 0);
            gemm(e_bf, 256, SE, 0, bpu + (long)i * 8192, 256, 0, 0, pu_b + (long)i * 32, nullptr, peb, 32, 8192, 0, 256, 32, 256, Bc, 1, 1.f, 0, 0);
            k_abt<<<dim3(4, 4, Bc), 256, 0, stream>>>(
                pub, 32, 8192, 0, peb, 32, 8192, 0, pair, 256, SE, 0,
                nullptr, 0, 0, 0, nullptr, nullptr, 0, pair_w, i, nullptr, nullptr,
                p_ulen, p_elen, 256, 256, 1, 1);
        }
    }

    // ===== output head =====
    gemm(u_bf, 256, SE, 0, bso, 256, 0, 0, so_b, nullptr, sub_, 32, 8192, 0, 256, 32, 256, Bc, 1, 1.f, 0, 0);
    gemm(e_bf, 256, SE, 0, bso, 256, 0, 0, so_b, nullptr, seb, 32, 8192, 0, 256, 32, 256, Bc, 1, 1.f, 0, 0);
    // gate = mask * sigmoid(so_bpp_w*pair + so_bpp_b + s_u.s_e^T)   (reuse scores buf)
    k_abt<<<dim3(4, 4, Bc), 256, 0, stream>>>(
        sub_, 32, 8192, 0, seb, 32, 8192, 0, scores, 256, SE, 0,
        pair, 256, SE, 0, nullptr, nullptr, 0, nullptr, 0, so_bw, so_bb,
        p_ulen, p_elen, 256, 256, 1, 2);
    k_rowcol<<<Bc, 256, 0, stream>>>(scores, gu, ge);
    k_gatevec<<<dim3(Bc * 256, 2), 256, 0, stream>>>(u_enc, e_enc, gate_w, gate_b, p_ulen, p_elen, ugv, egv);
    k_wsum<<<Bc, 256, 0, stream>>>(gu, u_enc, fv, 0);
    k_wsum<<<Bc, 256, 0, stream>>>(ge, e_enc, fv, 256);
    k_wsum<<<Bc, 256, 0, stream>>>(ugv, u_enc, fv, 512);
    k_wsum<<<Bc, 256, 0, stream>>>(egv, e_enc, fv, 768);
    k_tail<<<Bc, 64, 0, stream>>>(in_oh, in_he, fv);
    k_head<<<Bc, 256, 0, stream>>>(fv, on_s, on_b, od_w, od_b, ow_w, ow_b, (float*)d_out);
}
